// FixedLengthDemonstrationAttentionQModel_64175401337270
// MI455X (gfx1250) — matrix operand order repaired and checked by reading, not compiled, not run on hardware
//
#include <hip/hip_runtime.h>
#include <hip/hip_bf16.h>
#include <stdint.h>

// Problem constants
#define TT 64
#define BB 256
#define DD 512
#define LL 50
#define AA 18
#define HH 32
#define MTOT (TT * BB)          // 16384 rows of state
#define NFUSE 82                // 50 attn logits + 32 curX columns
#define NPAD 96                 // padded to 6 tiles of 16
#define NTILES 6

typedef __attribute__((ext_vector_type(16))) __bf16 v16bf;
typedef __attribute__((ext_vector_type(8)))  __bf16 v8bf;
typedef __attribute__((ext_vector_type(8)))  float  v8f;

// ---------------------------------------------------------------------------
// Pack kernel A: fused, zero-padded bf16 weight matrix Wf[96][512] + bias[96].
//   n <  50 : Wf[n][k] = W_attn[n][k]          (cur part of attn),  bias=b_attn
//   50..81  : Wf[n][k] = W_comb[n-50][512+k]   (cur part of comb),  bias=b_comb
//   82..95  : zeros
// ---------------------------------------------------------------------------
__global__ __launch_bounds__(256) void pack_w_kernel(
    const float* __restrict__ W_attn,   // [50, 544]
    const float* __restrict__ b_attn,   // [50]
    const float* __restrict__ W_comb,   // [32, 1024]
    const float* __restrict__ b_comb,   // [32]
    __bf16* __restrict__ Wf,            // [96, 512]
    float* __restrict__ biasF)          // [96]
{
    const int idx = blockIdx.x * 256 + threadIdx.x;   // grid 192 x 256 = 96*512
    const int n = idx >> 9;
    const int k = idx & 511;
    float v = 0.0f;
    if (n < LL)          v = W_attn[(size_t)n * (DD + HH) + k];
    else if (n < NFUSE)  v = W_comb[(size_t)(n - LL) * (2 * DD) + DD + k];
    Wf[idx] = (__bf16)v;

    if (idx < NPAD) {
        float bv = 0.0f;
        if (idx < LL)         bv = b_attn[idx];
        else if (idx < NFUSE) bv = b_comb[idx - LL];
        biasF[idx] = bv;
    }
}

// ---------------------------------------------------------------------------
// Pack kernel B: convert state [16384,512] f32 -> bf16 (8 elements/thread).
// ---------------------------------------------------------------------------
__global__ __launch_bounds__(256) void cvt_state_kernel(
    const float* __restrict__ state,
    __bf16* __restrict__ stateBf)
{
    const size_t base = ((size_t)blockIdx.x * 256 + threadIdx.x) * 8;  // grid 4096
    const float4 f0 = *(const float4*)(state + base);
    const float4 f1 = *(const float4*)(state + base + 4);
    v8bf o;
    o[0] = (__bf16)f0.x; o[1] = (__bf16)f0.y; o[2] = (__bf16)f0.z; o[3] = (__bf16)f0.w;
    o[4] = (__bf16)f1.x; o[5] = (__bf16)f1.y; o[6] = (__bf16)f1.z; o[7] = (__bf16)f1.w;
    *(v8bf*)(stateBf + base) = o;
}

// ---------------------------------------------------------------------------
// Kernel 1: fused h-independent projections of state, bf16 WMMA, f32 accum.
// GEMM: M=16384, N=96 (82 live), K=512, k-step 32. Branch-free vector loads.
// Grid: 256 blocks x 128 threads (4 waves); wave w owns M tile mBase..+15 and
// accumulates all 6 N tiles across the K loop.
// ---------------------------------------------------------------------------
__global__ __launch_bounds__(128) void proj_kernel(
    const __bf16* __restrict__ stateBf, // [16384, 512] bf16
    const __bf16* __restrict__ Wf,      // [96, 512]    bf16 (zero padded)
    const float*  __restrict__ biasF,   // [96]
    float* __restrict__ logitsD,        // [16384, 50]
    float* __restrict__ curX)           // [16384, 32]
{
    const int wave = threadIdx.x >> 5;
    const int lane = threadIdx.x & 31;
    const int half = lane >> 4;         // 0: lanes 0-15, 1: lanes 16-31
    const int r    = lane & 15;
    const int mBase = blockIdx.x * 64 + wave * 16;

    const __bf16* arow = stateBf + (size_t)(mBase + r) * DD;
    const __bf16* brow = Wf + (size_t)r * DD;   // row n = j*16 + r; step 16*512 per j

    v8f acc[NTILES] = {};

    for (int k0 = 0; k0 < DD; k0 += 32) {
        // A fragment (row M=r, contiguous K to match B):
        //   lane<16 : e0..15 -> K=k0+0..15
        //   lane>=16: e0..15 -> K=k0+16..31
        const v16bf a = *(const v16bf*)(arow + k0 + half * 16);

#pragma unroll
        for (int j = 0; j < NTILES; ++j) {
            // B fragment (32x16 KxN): lane<16 -> K=k0..k0+15 of column N=r,
            // lane>=16 -> K=k0+16..k0+31. B[k][n] = Wf[n][k], 32B contiguous.
            const v16bf bf = *(const v16bf*)(brow + (size_t)j * 16 * DD + k0 + half * 16);
            acc[j] = __builtin_amdgcn_wmma_f32_16x16x32_bf16(
                         false, a, false, bf, (short)0, acc[j], false, false);
        }
    }

    // Epilogue. C/D layout: vgpr v, lanes 0-15 -> (M=v, N=lane),
    // lanes 16-31 -> (M=v+8, N=lane-16).
#pragma unroll
    for (int j = 0; j < NTILES; ++j) {
        const int n = j * 16 + r;
        const float bias = biasF[n];
#pragma unroll
        for (int v = 0; v < 8; ++v) {
            const int m = mBase + half * 8 + v;
            const float val = acc[j][v] + bias;
            if (n < LL)
                logitsD[(size_t)m * LL + n] = val;
            else if (n < NFUSE)
                curX[(size_t)m * HH + (n - LL)] = val;
        }
    }
}

// ---------------------------------------------------------------------------
// Async stream of one demo tile (50x512 f32 = 102400 B) into LDS.
// 256 threads x float4 x 25 rounds. Uses CDNA5 async-to-LDS path (ASYNCcnt).
// ---------------------------------------------------------------------------
__device__ __forceinline__ void tile_async(const float* __restrict__ gsrc,
                                           float* ldst, int tid)
{
    // Generic shared pointer: low 32 bits are the LDS byte offset (ISA 10.2).
    uint32_t l = (uint32_t)(uintptr_t)ldst + (uint32_t)tid * 16u;
    uint64_t g = (uint64_t)(uintptr_t)(gsrc + tid * 4);
#pragma unroll
    for (int rr = 0; rr < 25; ++rr) {
        asm volatile("global_load_async_to_lds_b128 %0, %1, off"
                     :: "v"(l), "v"(g) : "memory");
        l += 256u * 16u;
        g += 256ull * 16ull;
    }
}

__device__ __forceinline__ float sigmoidf_(float x) {
    return 1.0f / (1.0f + __expf(-x));
}

// ---------------------------------------------------------------------------
// Kernel 2: sequential scan. One workgroup per batch element b (grid=256),
// 256 threads (8 waves). Demo tile double-buffered in LDS via async loads.
// ---------------------------------------------------------------------------
__global__ __launch_bounds__(256) void recur_kernel(
    const float* __restrict__ demo,     // [T,B,L,D]
    const float* __restrict__ h0,       // [B,H]
    const float* __restrict__ c0,       // [B,H]
    const float* __restrict__ W_attn,   // [50, 544] (cols 512.. = h part)
    const float* __restrict__ W_comb,   // [32, 1024] (cols 0..511 = applied)
    const float* __restrict__ W_ih, const float* __restrict__ b_ih,   // [128,32],[128]
    const float* __restrict__ W_hh, const float* __restrict__ b_hh,   // [128,32],[128]
    const float* __restrict__ W_mid, const float* __restrict__ b_mid, // [64,32],[64]
    const float* __restrict__ W_out, const float* __restrict__ b_out, // [18,64],[18]
    const float* __restrict__ logitsD,  // [T*B, 50]
    const float* __restrict__ curX,     // [T*B, 32]
    float* __restrict__ q,              // [T,B,A]
    float* __restrict__ hT,             // [B,H]
    float* __restrict__ cT)             // [B,H]
{
    __shared__ float sdemo[2][LL * DD];   // 204800 B double buffer
    __shared__ float sh[HH], sc[HH];
    __shared__ float sLog[LL];
    __shared__ float sw[LL];
    __shared__ float sApplied[DD];
    __shared__ float sPart[256];
    __shared__ float sx[HH];
    __shared__ float sg[4 * HH];
    __shared__ float smid[2 * HH];
    __shared__ float sred[1];

    const int b   = blockIdx.x;
    const int tid = threadIdx.x;

    if (tid < HH) {
        sh[tid] = h0[b * HH + tid];
        sc[tid] = c0[b * HH + tid];
    }

    // Prefetch tile t=0.
    tile_async(demo + ((size_t)0 * BB + b) * (size_t)(LL * DD), &sdemo[0][0], tid);

    for (int t = 0; t < TT; ++t) {
        const int cur = t & 1;

        // Drain this wave's async loads for tile t, then make LDS visible WG-wide.
        asm volatile("s_wait_asynccnt 0" ::: "memory");
        __syncthreads();

        // Kick off the prefetch of tile t+1 into the other buffer.
        if (t + 1 < TT)
            tile_async(demo + ((size_t)(t + 1) * BB + b) * (size_t)(LL * DD),
                       &sdemo[(t + 1) & 1][0], tid);

        // ---- attention logits: logitsD + h . W_attn[:, 512:544] ----
        if (tid < LL) {
            const float* wa = W_attn + (size_t)tid * (DD + HH) + DD;
            float acc = logitsD[((size_t)t * BB + b) * LL + tid];
#pragma unroll
            for (int j = 0; j < HH; ++j) acc += sh[j] * wa[j];
            sLog[tid] = acc;
        }
        __syncthreads();

        // ---- softmax over L=50 (serial in one lane: trivial cost) ----
        if (tid == 0) {
            float mx = sLog[0];
            for (int l = 1; l < LL; ++l) mx = fmaxf(mx, sLog[l]);
            float s = 0.0f;
            for (int l = 0; l < LL; ++l) { float e = __expf(sLog[l] - mx); sw[l] = e; s += e; }
            sred[0] = 1.0f / s;
        }
        __syncthreads();
        const float inv = sred[0];

        // ---- applied[d] = (1/Z) * sum_l w_l * demo[t,b,l,d] ----
#pragma unroll
        for (int dd = 0; dd < 2; ++dd) {
            const int d = tid + dd * 256;
            float acc = 0.0f;
            for (int l = 0; l < LL; ++l) acc += sw[l] * sdemo[cur][l * DD + d];
            sApplied[d] = acc * inv;
        }
        __syncthreads();

        // ---- x = relu(applied . W_comb[:,0:512]^T + curX) ----
        {
            const int hh   = tid >> 3;   // 0..31
            const int part = tid & 7;    // 0..7, each sums 64 elems
            const float* wc = W_comb + (size_t)hh * (2 * DD) + part * 64;
            const float* ap = sApplied + part * 64;
            float acc = 0.0f;
#pragma unroll 8
            for (int j = 0; j < 64; ++j) acc += ap[j] * wc[j];
            sPart[tid] = acc;
        }
        __syncthreads();
        if (tid < HH) {
            float acc = curX[((size_t)t * BB + b) * HH + tid];
#pragma unroll
            for (int p = 0; p < 8; ++p) acc += sPart[tid * 8 + p];
            sx[tid] = fmaxf(acc, 0.0f);
        }
        __syncthreads();

        // ---- LSTM gates: x@W_ih^T + b_ih + h@W_hh^T + b_hh ----
        if (tid < 4 * HH) {
            const float* wi = W_ih + (size_t)tid * HH;
            const float* wh = W_hh + (size_t)tid * HH;
            float acc = b_ih[tid] + b_hh[tid];
#pragma unroll
            for (int j = 0; j < HH; ++j) acc += sx[j] * wi[j] + sh[j] * wh[j];
            sg[tid] = acc;
        }
        __syncthreads();

        // ---- LSTM state update (gate order i, f, g, o) ----
        if (tid < HH) {
            const float ig = sigmoidf_(sg[tid]);
            const float fg = sigmoidf_(sg[HH + tid]);
            const float gg = tanhf(sg[2 * HH + tid]);
            const float og = sigmoidf_(sg[3 * HH + tid]);
            const float cn = fg * sc[tid] + ig * gg;
            sc[tid] = cn;
            sh[tid] = og * tanhf(cn);
        }
        __syncthreads();

        // ---- mid = h_new @ W_mid^T + b_mid ----
        if (tid < 2 * HH) {
            const float* wm = W_mid + (size_t)tid * HH;
            float acc = b_mid[tid];
#pragma unroll
            for (int j = 0; j < HH; ++j) acc += sh[j] * wm[j];
            smid[tid] = acc;
        }
        __syncthreads();

        // ---- q = mid @ W_out^T + b_out ----
        if (tid < AA) {
            const float* wo = W_out + (size_t)tid * (2 * HH);
            float acc = b_out[tid];
#pragma unroll
            for (int j = 0; j < 2 * HH; ++j) acc += smid[j] * wo[j];
            q[((size_t)t * BB + b) * AA + tid] = acc;
        }
        __syncthreads();   // protect sLog/sw/sh reuse next iteration
    }

    if (tid < HH) {
        hT[b * HH + tid] = sh[tid];
        cT[b * HH + tid] = sc[tid];
    }
}

// ---------------------------------------------------------------------------
extern "C" void kernel_launch(void* const* d_in, const int* in_sizes, int n_in,
                              void* d_out, int out_size, void* d_ws, size_t ws_size,
                              hipStream_t stream)
{
    (void)in_sizes; (void)n_in; (void)out_size; (void)ws_size;

    const float* state  = (const float*)d_in[0];
    const float* demo   = (const float*)d_in[1];
    const float* h0     = (const float*)d_in[2];
    const float* c0     = (const float*)d_in[3];
    const float* W_attn = (const float*)d_in[4];
    const float* b_attn = (const float*)d_in[5];
    const float* W_comb = (const float*)d_in[6];
    const float* b_comb = (const float*)d_in[7];
    const float* W_ih   = (const float*)d_in[8];
    const float* b_ih   = (const float*)d_in[9];
    const float* W_hh   = (const float*)d_in[10];
    const float* b_hh   = (const float*)d_in[11];
    const float* W_mid  = (const float*)d_in[12];
    const float* b_mid  = (const float*)d_in[13];
    const float* W_out  = (const float*)d_in[14];
    const float* b_out  = (const float*)d_in[15];

    float* out = (float*)d_out;
    float* q  = out;                               // [T,B,A]
    float* hT = out + (size_t)TT * BB * AA;        // [B,H]
    float* cT = hT + (size_t)BB * HH;              // [B,H]

    // Workspace layout (all 16B aligned):
    char* ws = (char*)d_ws;
    float*  logitsD = (float*)ws;                                   // 16384*50 f32
    float*  curX    = logitsD + (size_t)MTOT * LL;                  // 16384*32 f32
    __bf16* stateBf = (__bf16*)(curX + (size_t)MTOT * HH);          // 16384*512 bf16
    __bf16* WfBf    = stateBf + (size_t)MTOT * DD;                  // 96*512 bf16
    float*  biasF   = (float*)(WfBf + (size_t)NPAD * DD);           // 96 f32

    pack_w_kernel<<<(NPAD * DD) / 256, 256, 0, stream>>>(
        W_attn, b_attn, W_comb, b_comb, WfBf, biasF);

    cvt_state_kernel<<<(MTOT * DD) / (256 * 8), 256, 0, stream>>>(state, stateBf);

    proj_kernel<<<MTOT / 64, 128, 0, stream>>>(
        stateBf, WfBf, biasF, logitsD, curX);

    recur_kernel<<<BB, 256, 0, stream>>>(
        demo, h0, c0, W_attn, W_comb,
        W_ih, b_ih, W_hh, b_hh, W_mid, b_mid, W_out, b_out,
        logitsD, curX, q, hT, cT);
}
